// ReferenceFFN_38242388803681
// MI455X (gfx1250) — matrix operand order repaired and checked by reading, not compiled, not run on hardware
//
#include <hip/hip_runtime.h>
#include <hip/hip_bf16.h>
#include <math.h>

// ---------------------------------------------------------------------------
// Sparse (top-k) SwiGLU FFN for MI455X (gfx1250), wave32 + bf16 WMMA +
// CDNA5 async global->LDS staging (GLOBAL_LOAD_ASYNC_TO_LDS_B128, ASYNCcnt).
//
//  K0 : pre-convert x -> bf16 ; w_gate/w_up/w_down -> bf16 transposed [n][k]
//  K1 : fused dual GEMM  G = x@w_gate (f32), U = x@w_up (bf16)
//       (LDS double-buffered via async DMA, v_wmma_f32_16x16x32_bf16)
//  K2 : per-row 4-pass radix select of 512th-largest G -> thresh[row]
//  K2b: Z = (G >= thresh) ? silu(G)*U : 0   (overwrites U in place, bf16)
//  K3 : out = Z @ w_down (async double-buffered bf16 WMMA GEMM)
// ---------------------------------------------------------------------------

#define D_MODEL 2048
#define D_FFN   8192
#define M_TOK   8192
#define TOP_K   512

#define TILE_M 128
#define TILE_N 128
#define TILE_K 32
#define KP     40   // TILE_K + 8 bf16 pad -> 80B row stride (16B aligned)

typedef __attribute__((ext_vector_type(8)))  __bf16 v8bf;
typedef __attribute__((ext_vector_type(16))) __bf16 v16bf;
typedef __attribute__((ext_vector_type(8)))  float  v8f;

__device__ __forceinline__ unsigned short f2bf_bits(float f) {
  unsigned u = __builtin_bit_cast(unsigned, f);
  u += 0x7FFFu + ((u >> 16) & 1u);            // round-to-nearest-even
  return (unsigned short)(u >> 16);
}
__device__ __forceinline__ float bf2f(unsigned short h) {
  unsigned u = ((unsigned)h) << 16;
  return __builtin_bit_cast(float, u);
}

__device__ __forceinline__ v8f wmma_bf16(v16bf a, v16bf b, v8f c) {
  return __builtin_amdgcn_wmma_f32_16x16x32_bf16(
      false, a, false, b, (short)0, c, false, false);
}

// --- CDNA5 async DMA: global -> LDS, 16B per lane, tracked by ASYNCcnt -----
__device__ __forceinline__ unsigned lds_off_u32(const void* p) {
  // LDS aperture occupies the high 32 bits of a generic address
  // (ISA 10.2 aperture check), so low 32 bits == wave-relative LDS offset.
  return (unsigned)(unsigned long long)p;
}
__device__ __forceinline__ void async_b128(unsigned ldsoff, unsigned voff,
                                           const void* sbase) {
  asm volatile("global_load_async_to_lds_b128 %0, %1, %2"
               :: "v"(ldsoff), "v"(voff), "s"(sbase)
               : "memory");
}
#define WAIT_ASYNC(n) asm volatile("s_wait_asynccnt " #n ::: "memory")

// Stage a 128-row x 32-col bf16 tile (64B/row) into LDS (stride KP bf16).
// 512 x 16B chunks, 2 async issues per thread.
__device__ __forceinline__ void stage_tile_async(
    const __bf16* ldsBase, const unsigned short* src, int row0, int k0,
    int strideElems, int tid) {
  unsigned base = lds_off_u32(ldsBase);
#pragma unroll
  for (int it = 0; it < 2; ++it) {
    int c = tid + (it << 8);            // 0..511
    int r = c >> 2;                     // tile row 0..127
    int j = c & 3;                      // 16B chunk in 64B row
    unsigned loff = base + (unsigned)(r * (KP * 2)) + (unsigned)(j << 4);
    unsigned voff = (unsigned)(((size_t)(row0 + r) * strideElems + k0) * 2)
                    + (unsigned)(j << 4);
    async_b128(loff, voff, src);
  }
}

// --- WMMA fragment loads (ISA 7.12.2 bf16 layouts) -------------------------
__device__ __forceinline__ v16bf load_a_frag(const __bf16* lds, int rowBase, int lane) {
  int row = rowBase + (lane & 15);
  const __bf16* p = lds + row * KP + ((lane >> 4) << 3);   // +0 / +8
  v8bf lo = *(const v8bf*)p;
  v8bf hi = *(const v8bf*)(p + 16);
  return __builtin_shufflevector(lo, hi, 0,1,2,3,4,5,6,7,8,9,10,11,12,13,14,15);
}
__device__ __forceinline__ v16bf load_b_frag(const __bf16* lds, int colBase, int lane) {
  int col = colBase + (lane & 15);
  const __bf16* p = lds + col * KP + ((lane >> 4) << 3);   // +0 / +8
  v8bf lo = *(const v8bf*)p;
  v8bf hi = *(const v8bf*)(p + 16);
  return __builtin_shufflevector(lo, hi, 0,1,2,3,4,5,6,7,8,9,10,11,12,13,14,15);
}

// ---------------------------------------------------------------------------
// K0a: elementwise f32 -> bf16 (x matrix; layout preserved)
// ---------------------------------------------------------------------------
__global__ __launch_bounds__(256) void cvt_bf16_kernel(
    const float* __restrict__ src, unsigned short* __restrict__ dst) {
  size_t base = ((size_t)blockIdx.x * 256 + threadIdx.x) * 4;
  float4 v = *(const float4*)(src + base);
  unsigned long long pk =
      (unsigned long long)f2bf_bits(v.x)
    | ((unsigned long long)f2bf_bits(v.y) << 16)
    | ((unsigned long long)f2bf_bits(v.z) << 32)
    | ((unsigned long long)f2bf_bits(v.w) << 48);
  *(unsigned long long*)(dst + base) = pk;
}

// ---------------------------------------------------------------------------
// K0b: f32 [K][N] -> bf16 [N][K] (transpose-convert via LDS tile)
// ---------------------------------------------------------------------------
__global__ __launch_bounds__(256) void transpose_bf16_kernel(
    const float* __restrict__ src, unsigned short* __restrict__ dst,
    int K, int N) {
  __shared__ float t[64][65];
  const int kb = blockIdx.y << 6, nb = blockIdx.x << 6;
  for (int e = threadIdx.x; e < 4096; e += 256) {
    int r = e >> 6, c = e & 63;
    t[r][c] = src[(size_t)(kb + r) * N + nb + c];
  }
  __syncthreads();
  for (int e = threadIdx.x; e < 4096; e += 256) {
    int r = e >> 6, c = e & 63;          // r: n-index, c: k-index
    dst[(size_t)(nb + r) * K + kb + c] = f2bf_bits(t[c][r]);
  }
}

// ---------------------------------------------------------------------------
// K1: fused dual GEMM.  Xb[M][2048], Wgb/Wub[8192][2048] (all bf16, [n][k]).
// Block 128x128, 8 waves 2x4 -> 64x32/wave/matrix. Async double-buffered.
// ---------------------------------------------------------------------------
__global__ __launch_bounds__(256) void gemm_gu_kernel(
    const unsigned short* __restrict__ Xb, const unsigned short* __restrict__ Wgb,
    const unsigned short* __restrict__ Wub, float* __restrict__ G,
    unsigned short* __restrict__ U) {
  __shared__ __align__(16) __bf16 lA [2][TILE_M * KP];
  __shared__ __align__(16) __bf16 lBg[2][TILE_N * KP];
  __shared__ __align__(16) __bf16 lBu[2][TILE_N * KP];

  const int tid  = threadIdx.x;
  const int lane = tid & 31;
  const int wave = tid >> 5;
  const int wm   = (wave >> 2) << 6;
  const int wn   = (wave & 3) << 5;
  const int m0   = blockIdx.y * TILE_M;
  const int n0   = blockIdx.x * TILE_N;

  v8f accG[4][2] = {};
  v8f accU[4][2] = {};

  stage_tile_async(lA[0],  Xb,  m0, 0, D_MODEL, tid);
  stage_tile_async(lBg[0], Wgb, n0, 0, D_MODEL, tid);
  stage_tile_async(lBu[0], Wub, n0, 0, D_MODEL, tid);

  const int KT = D_MODEL / TILE_K;      // 64
  for (int kt = 0; kt < KT; ++kt) {
    const int cur = kt & 1, nxt = cur ^ 1;
    if (kt + 1 < KT) {
      const int k0 = (kt + 1) * TILE_K;
      stage_tile_async(lA[nxt],  Xb,  m0, k0, D_MODEL, tid);
      stage_tile_async(lBg[nxt], Wgb, n0, k0, D_MODEL, tid);
      stage_tile_async(lBu[nxt], Wub, n0, k0, D_MODEL, tid);
      WAIT_ASYNC(6);                    // 6 newest = next tile; current done
    } else {
      WAIT_ASYNC(0);
    }
    __syncthreads();                    // current tile visible to all waves

    v16bf bg0 = load_b_frag(lBg[cur], wn,      lane);
    v16bf bg1 = load_b_frag(lBg[cur], wn + 16, lane);
    v16bf bu0 = load_b_frag(lBu[cur], wn,      lane);
    v16bf bu1 = load_b_frag(lBu[cur], wn + 16, lane);
#pragma unroll
    for (int mi = 0; mi < 4; ++mi) {
      v16bf a = load_a_frag(lA[cur], wm + (mi << 4), lane);
      accG[mi][0] = wmma_bf16(a, bg0, accG[mi][0]);
      accG[mi][1] = wmma_bf16(a, bg1, accG[mi][1]);
      accU[mi][0] = wmma_bf16(a, bu0, accU[mi][0]);
      accU[mi][1] = wmma_bf16(a, bu1, accU[mi][1]);
    }
    __syncthreads();                    // all reads done before buf reuse
  }

#pragma unroll
  for (int mi = 0; mi < 4; ++mi)
#pragma unroll
    for (int ni = 0; ni < 2; ++ni) {
      int n = n0 + wn + (ni << 4) + (lane & 15);
      int m = m0 + wm + (mi << 4) + ((lane >> 4) << 3);
      float*          gp = G + (size_t)m * D_FFN + n;
      unsigned short* up = U + (size_t)m * D_FFN + n;
#pragma unroll
      for (int v = 0; v < 8; ++v) {
        gp[(size_t)v * D_FFN] = accG[mi][ni][v];
        up[(size_t)v * D_FFN] = f2bf_bits(accU[mi][ni][v]);
      }
    }
}

// ---------------------------------------------------------------------------
// K2: per-row 512th-largest of 8192 via 4-pass MSB radix select (LDS-cached).
// ---------------------------------------------------------------------------
__device__ __forceinline__ unsigned fkey(float f) {   // monotone f32 -> u32
  unsigned u = __builtin_bit_cast(unsigned, f);
  return (u & 0x80000000u) ? ~u : (u | 0x80000000u);
}
__device__ __forceinline__ float key2f(unsigned k) {
  unsigned u = (k & 0x80000000u) ? (k & 0x7FFFFFFFu) : ~k;
  return __builtin_bit_cast(float, u);
}

__global__ __launch_bounds__(256) void topk_threshold_kernel(
    const float* __restrict__ G, float* __restrict__ thresh) {
  __shared__ unsigned skey[D_FFN];      // 32 KB row cache
  __shared__ unsigned hist[256];
  __shared__ unsigned s_sel;
  __shared__ int      s_k;

  const int row = blockIdx.x;
  const float* g = G + (size_t)row * D_FFN;
  for (int i = threadIdx.x; i < D_FFN; i += 256) skey[i] = fkey(g[i]);
  __syncthreads();

  unsigned prefix = 0, pmask = 0;
  int k = TOP_K;
  for (int shift = 24; shift >= 0; shift -= 8) {
    hist[threadIdx.x] = 0;
    __syncthreads();
    for (int i = threadIdx.x; i < D_FFN; i += 256) {
      unsigned key = skey[i];
      if ((key & pmask) == prefix) atomicAdd(&hist[(key >> shift) & 255u], 1u);
    }
    __syncthreads();
    if (threadIdx.x == 0) {
      int kk = k, b = 255;
      for (; b > 0; --b) {
        int c = (int)hist[b];
        if (kk <= c) break;
        kk -= c;
      }
      s_sel = (unsigned)b;
      s_k   = kk;
    }
    __syncthreads();
    prefix |= (s_sel << shift);
    pmask  |= (255u << shift);
    k = s_k;
    __syncthreads();
  }
  if (threadIdx.x == 0) thresh[row] = key2f(prefix);  // exact 512th largest
}

// ---------------------------------------------------------------------------
// K2b: Z = (G >= thresh[row]) ? silu(G)*U : 0, overwriting U in place (bf16).
// ---------------------------------------------------------------------------
__global__ __launch_bounds__(256) void zmask_kernel(
    const float* __restrict__ G, unsigned short* __restrict__ UZ,
    const float* __restrict__ thresh) {
  size_t base = ((size_t)blockIdx.x * 256 + threadIdx.x) * 4;
  const int row = (int)(base >> 13);        // uniform per block (8192/row)
  const float t = thresh[row];
  float4 g4 = *(const float4*)(G + base);
  unsigned short* up = UZ + base;
  float gv[4] = {g4.x, g4.y, g4.z, g4.w};
  unsigned short zz[4];
#pragma unroll
  for (int j = 0; j < 4; ++j) {
    float g = gv[j];
    zz[j] = (g >= t) ? f2bf_bits((g / (1.0f + __expf(-g))) * bf2f(up[j]))
                     : (unsigned short)0;
  }
  unsigned long long pk =
      (unsigned long long)zz[0] | ((unsigned long long)zz[1] << 16)
    | ((unsigned long long)zz[2] << 32) | ((unsigned long long)zz[3] << 48);
  *(unsigned long long*)up = pk;
}

// ---------------------------------------------------------------------------
// K3: out = Z @ w_down.  Zb[M][8192], Wdb[2048][8192] (bf16 [n][k]).
// ---------------------------------------------------------------------------
__global__ __launch_bounds__(256) void gemm_down_kernel(
    const unsigned short* __restrict__ Zb, const unsigned short* __restrict__ Wdb,
    float* __restrict__ Out) {
  __shared__ __align__(16) __bf16 lA[2][TILE_M * KP];
  __shared__ __align__(16) __bf16 lB[2][TILE_N * KP];

  const int tid  = threadIdx.x;
  const int lane = tid & 31;
  const int wave = tid >> 5;
  const int wm   = (wave >> 2) << 6;
  const int wn   = (wave & 3) << 5;
  const int m0   = blockIdx.y * TILE_M;
  const int n0   = blockIdx.x * TILE_N;

  v8f acc[4][2] = {};

  stage_tile_async(lA[0], Zb,  m0, 0, D_FFN, tid);
  stage_tile_async(lB[0], Wdb, n0, 0, D_FFN, tid);

  const int KT = D_FFN / TILE_K;        // 256
  for (int kt = 0; kt < KT; ++kt) {
    const int cur = kt & 1, nxt = cur ^ 1;
    if (kt + 1 < KT) {
      const int k0 = (kt + 1) * TILE_K;
      stage_tile_async(lA[nxt], Zb,  m0, k0, D_FFN, tid);
      stage_tile_async(lB[nxt], Wdb, n0, k0, D_FFN, tid);
      WAIT_ASYNC(4);
    } else {
      WAIT_ASYNC(0);
    }
    __syncthreads();

    v16bf b0 = load_b_frag(lB[cur], wn,      lane);
    v16bf b1 = load_b_frag(lB[cur], wn + 16, lane);
#pragma unroll
    for (int mi = 0; mi < 4; ++mi) {
      v16bf a = load_a_frag(lA[cur], wm + (mi << 4), lane);
      acc[mi][0] = wmma_bf16(a, b0, acc[mi][0]);
      acc[mi][1] = wmma_bf16(a, b1, acc[mi][1]);
    }
    __syncthreads();
  }

#pragma unroll
  for (int mi = 0; mi < 4; ++mi)
#pragma unroll
    for (int ni = 0; ni < 2; ++ni) {
      int n = n0 + wn + (ni << 4) + (lane & 15);
      int m = m0 + wm + (mi << 4) + ((lane >> 4) << 3);
      float* op = Out + (size_t)m * D_MODEL + n;
#pragma unroll
      for (int v = 0; v < 8; ++v) op[(size_t)v * D_MODEL] = acc[mi][ni][v];
    }
}

// ---------------------------------------------------------------------------
extern "C" void kernel_launch(void* const* d_in, const int* in_sizes, int n_in,
                              void* d_out, int out_size, void* d_ws, size_t ws_size,
                              hipStream_t stream) {
  (void)in_sizes; (void)n_in; (void)out_size; (void)ws_size;
  const float* x      = (const float*)d_in[0];
  const float* w_gate = (const float*)d_in[1];
  const float* w_up   = (const float*)d_in[2];
  const float* w_down = (const float*)d_in[3];
  float* out = (float*)d_out;

  char* ws = (char*)d_ws;
  float*          G   = (float*)(ws);                            // 256 MiB
  unsigned short* U   = (unsigned short*)(ws + 268435456ull);    // 128 MiB
  unsigned short* Xb  = (unsigned short*)(ws + 402653184ull);    //  32 MiB
  unsigned short* Wgb = (unsigned short*)(ws + 436207616ull);    //  32 MiB
  unsigned short* Wub = (unsigned short*)(ws + 469762048ull);    //  32 MiB
  unsigned short* Wdb = (unsigned short*)(ws + 503316480ull);    //  32 MiB
  float*          th  = (float*)(ws + 536870912ull);             //  32 KiB

  // K0: pre-convert operands to bf16 (weights transposed to [n][k])
  cvt_bf16_kernel<<<(M_TOK * D_MODEL) / 1024, 256, 0, stream>>>(x, Xb);
  transpose_bf16_kernel<<<dim3(D_FFN / 64, D_MODEL / 64), 256, 0, stream>>>(
      w_gate, Wgb, D_MODEL, D_FFN);
  transpose_bf16_kernel<<<dim3(D_FFN / 64, D_MODEL / 64), 256, 0, stream>>>(
      w_up, Wub, D_MODEL, D_FFN);
  transpose_bf16_kernel<<<dim3(D_MODEL / 64, D_FFN / 64), 256, 0, stream>>>(
      w_down, Wdb, D_FFN, D_MODEL);

  gemm_gu_kernel<<<dim3(D_FFN / TILE_N, M_TOK / TILE_M), 256, 0, stream>>>(
      Xb, Wgb, Wub, G, U);
  topk_threshold_kernel<<<M_TOK, 256, 0, stream>>>(G, th);
  zmask_kernel<<<(M_TOK * D_FFN) / 1024, 256, 0, stream>>>(G, U, th);
  gemm_down_kernel<<<dim3(D_MODEL / TILE_N, M_TOK / TILE_M), 256, 0, stream>>>(
      U, Wdb, out);
}
